// AtrousIIBlock_73555609911910
// MI455X (gfx1250) — matrix operand order repaired and checked by reading, not compiled, not run on hardware
//
#include <hip/hip_runtime.h>
#include <hip/hip_bf16.h>

// ---------------------------------------------------------------------------
// Types for CDNA5 WMMA (wave32): bf16 A/B fragments, f32 accumulator.
// ---------------------------------------------------------------------------
typedef __bf16 bf16_t;
typedef bf16_t v16bf __attribute__((ext_vector_type(16)));
typedef float  v8f   __attribute__((ext_vector_type(8)));
typedef float  v4f   __attribute__((ext_vector_type(4)));
typedef unsigned int v4u __attribute__((ext_vector_type(4)));

union Frag16 { v16bf v; v4u q[2]; };
union BF4    { bf16_t h[4]; unsigned long long u; };

#define C_CH 64
#define TPW  8      // tiles (of 16 gather/scatter pairs) per wave

// ---------------------------------------------------------------------------
// Zero-fill (float4 stores), grid-stride.
// ---------------------------------------------------------------------------
__global__ void fill0_f4(float* __restrict__ p, long n4) {
    long i = (long)blockIdx.x * blockDim.x + threadIdx.x;
    long stride = (long)gridDim.x * blockDim.x;
    v4f z = {0.f, 0.f, 0.f, 0.f};
    for (; i < n4; i += stride) ((v4f*)p)[i] = z;
}

// ---------------------------------------------------------------------------
// fp32 -> bf16 conversion, 4 elements/thread.
// ---------------------------------------------------------------------------
__global__ void cvt_f32_bf16(const float* __restrict__ x, bf16_t* __restrict__ y, long n4) {
    long i = (long)blockIdx.x * blockDim.x + threadIdx.x;
    if (i >= n4) return;
    v4f d = ((const v4f*)x)[i];
    BF4 r;
    r.h[0] = (bf16_t)d.x; r.h[1] = (bf16_t)d.y; r.h[2] = (bf16_t)d.z; r.h[3] = (bf16_t)d.w;
    *(unsigned long long*)(y + 4 * i) = r.u;
}

// ---------------------------------------------------------------------------
// Pre-swizzle weights into WMMA B-fragment layout (bf16).
// frag layout: [k][ntile(4)][kblock(2)][lane(32)][j(16)]   (f = nt*2+kb)
//   cin  = kb*32 + (lane>>4)*16 + j
//   cout = nt*16 + (lane&15)
// ---------------------------------------------------------------------------
__global__ void wprep_frag(const float* __restrict__ W, bf16_t* __restrict__ frag, int n) {
    int i = blockIdx.x * blockDim.x + threadIdx.x;
    if (i >= n) return;
    int j    = i & 15;
    int lane = (i >> 4) & 31;
    int kb   = (i >> 9) & 1;
    int nt   = (i >> 10) & 3;
    int k    = i >> 12;
    int cin  = kb * 32 + (lane >> 4) * 16 + j;
    int cout = nt * 16 + (lane & 15);
    frag[i] = (bf16_t)W[((size_t)k * C_CH + cin) * C_CH + cout];
}

// ---------------------------------------------------------------------------
// Sparse conv, weight-stationary:
//   grid = (ceil(TPK / (4*TPW)), K) -- no integer division anywhere.
//   Each wave keeps the 8 B fragments of its offset k in VGPRs and sweeps
//   TPW consecutive 16-pair tiles:
//     gather 16 rows of xb -> two A fragments -> 8x v_wmma_f32_16x16x32_bf16
//     -> scatter 16x64 f32 via global_atomic_add_f32 (L2-resident).
//   Pads (out_idx == N) are packed at each offset's tail, so the first padded
//   tile terminates the wave's whole sweep (break).
// ---------------------------------------------------------------------------
__global__ __launch_bounds__(128)
void sconv_wmma_bf16(const bf16_t* __restrict__ xb, const bf16_t* __restrict__ wf,
                     const int* __restrict__ inx, const int* __restrict__ outx,
                     float* __restrict__ acc, int N, int TPK) {
    const int wave = threadIdx.x >> 5;
    const int lane = threadIdx.x & 31;
    const int k    = blockIdx.y;
    const int tile0 = (blockIdx.x * 4 + wave) * TPW;
    if (tile0 >= TPK) return;

    const size_t mapBase = (size_t)k * N;
    const int half = lane >> 4;
    const int mcol = lane & 15;     // tile row owned for gather; also D column

    // B fragments for offset k, resident across the tile sweep
    const bf16_t* wk = wf + (size_t)k * 4096 + lane * 16;
    Frag16 B[8];
#pragma unroll
    for (int f = 0; f < 8; ++f) {   // f = nt*2 + kb
        B[f].q[0] = *(const v4u*)(wk + f * 512);
        B[f].q[1] = *(const v4u*)(wk + f * 512 + 8);
    }

    int tEnd = tile0 + TPW;
    if (tEnd > TPK) tEnd = TPK;

    for (int t = tile0; t < tEnd; ++t) {
        const int base = t * 16;
        if (outx[mapBase + base] >= N) break;     // pad tail reached

        const int mrow   = base + mcol;
        const int mclamp = mrow < N - 1 ? mrow : N - 1;
        const int src    = inx[mapBase + mclamp];
        const int dst    = (mrow < N) ? outx[mapBase + mrow] : N;

        // A fragments: contiguous K order (lane half: K 0-15 vs 16-31)
        const bf16_t* xr = xb + (size_t)src * C_CH;
        Frag16 A0, A1;
        A0.q[0] = *(const v4u*)(xr + half * 16);
        A0.q[1] = *(const v4u*)(xr + half * 16 + 8);
        A1.q[0] = *(const v4u*)(xr + 32 + half * 16);
        A1.q[1] = *(const v4u*)(xr + 32 + half * 16 + 8);

        v8f a0 = {0,0,0,0,0,0,0,0}, a1 = a0, a2 = a0, a3 = a0;
        a0 = __builtin_amdgcn_wmma_f32_16x16x32_bf16(false, A0.v, false, B[0].v, (short)0, a0, false, false);
        a0 = __builtin_amdgcn_wmma_f32_16x16x32_bf16(false, A1.v, false, B[1].v, (short)0, a0, false, false);
        a1 = __builtin_amdgcn_wmma_f32_16x16x32_bf16(false, A0.v, false, B[2].v, (short)0, a1, false, false);
        a1 = __builtin_amdgcn_wmma_f32_16x16x32_bf16(false, A1.v, false, B[3].v, (short)0, a1, false, false);
        a2 = __builtin_amdgcn_wmma_f32_16x16x32_bf16(false, A0.v, false, B[4].v, (short)0, a2, false, false);
        a2 = __builtin_amdgcn_wmma_f32_16x16x32_bf16(false, A1.v, false, B[5].v, (short)0, a2, false, false);
        a3 = __builtin_amdgcn_wmma_f32_16x16x32_bf16(false, A0.v, false, B[6].v, (short)0, a3, false, false);
        a3 = __builtin_amdgcn_wmma_f32_16x16x32_bf16(false, A1.v, false, B[7].v, (short)0, a3, false, false);

        // D layout: VGPR r holds row M = half*8 + r, column = lane&15.
        int dstr[8];
#pragma unroll
        for (int r = 0; r < 8; ++r) dstr[r] = __shfl(dst, half * 8 + r, 32);

#pragma unroll
        for (int r = 0; r < 8; ++r) {
            if (dstr[r] < N) {
                float* orow = acc + (size_t)dstr[r] * C_CH + mcol;
                atomicAdd(orow + 0,  a0[r]);
                atomicAdd(orow + 16, a1[r]);
                atomicAdd(orow + 32, a2[r]);
                atomicAdd(orow + 48, a3[r]);
            }
        }
    }
}

// ---------------------------------------------------------------------------
// Instance-norm stats: per-channel sum & sumsq -> atomic into stats[0:64]/[64:128]
// ---------------------------------------------------------------------------
__global__ void in_stats(const float* __restrict__ v, float* __restrict__ stats, int n) {
    __shared__ float s_sum[256];
    __shared__ float s_sq[256];
    int col = threadIdx.x & 63;
    int seg = threadIdx.x >> 6;            // 4 row-lanes per channel
    float s = 0.f, q = 0.f;
    for (long i = (long)blockIdx.x * 4 + seg; i < n; i += (long)gridDim.x * 4) {
        float x = v[i * C_CH + col];
        s += x; q += x * x;
    }
    s_sum[threadIdx.x] = s;
    s_sq[threadIdx.x]  = q;
    __syncthreads();
    if (seg == 0) {
#pragma unroll
        for (int r = 1; r < 4; ++r) { s += s_sum[col + r * 64]; q += s_sq[col + r * 64]; }
        atomicAdd(&stats[col], s);
        atomicAdd(&stats[64 + col], q);
    }
}

__global__ void in_finalize(const float* __restrict__ stats, float* __restrict__ sb, int n) {
    int c = threadIdx.x;
    if (c >= 64) return;
    float inv = 1.0f / (float)n;
    float m = stats[c] * inv;
    float var = stats[64 + c] * inv - m * m;
    float s = rsqrtf(var + 1e-5f);
    sb[c] = s;
    sb[64 + c] = -m * s;
}

// ---------------------------------------------------------------------------
// apply1: y = bf16(relu(norm(conv1)))   (input to conv2)
// ---------------------------------------------------------------------------
__global__ void apply_norm_relu_bf16(const float* __restrict__ v, const float* __restrict__ sb,
                                     bf16_t* __restrict__ y, long n4) {
    long i = (long)blockIdx.x * blockDim.x + threadIdx.x;
    if (i >= n4) return;
    int c4 = ((int)(i & 15)) * 4;
    v4f d = ((const v4f*)v)[i];
    BF4 r;
#pragma unroll
    for (int j = 0; j < 4; ++j) {
        float t = d[j] * sb[c4 + j] + sb[64 + c4 + j];
        t = t > 0.f ? t : 0.f;
        r.h[j] = (bf16_t)t;
    }
    *(unsigned long long*)(y + 4 * i) = r.u;
}

// ---------------------------------------------------------------------------
// apply2: out = relu(norm(conv2) + x)   (residual, fp32 output)
// ---------------------------------------------------------------------------
__global__ void apply_norm_res_relu(const float* __restrict__ v, const float* __restrict__ sb,
                                    const float* __restrict__ x, float* __restrict__ out, long n4) {
    long i = (long)blockIdx.x * blockDim.x + threadIdx.x;
    if (i >= n4) return;
    int c4 = ((int)(i & 15)) * 4;
    v4f d = ((const v4f*)v)[i];
    v4f xr = ((const v4f*)x)[i];
    v4f o;
#pragma unroll
    for (int j = 0; j < 4; ++j) {
        float t = d[j] * sb[c4 + j] + sb[64 + c4 + j] + xr[j];
        o[j] = t > 0.f ? t : 0.f;
    }
    ((v4f*)out)[i] = o;
}

// ---------------------------------------------------------------------------
// Host-side orchestration (graph-capture safe: kernels on `stream` only).
// ---------------------------------------------------------------------------
extern "C" void kernel_launch(void* const* d_in, const int* in_sizes, int n_in,
                              void* d_out, int out_size, void* d_ws, size_t ws_size,
                              hipStream_t stream) {
    const float* x   = (const float*)d_in[0];
    const float* W1  = (const float*)d_in[1];
    const float* W2  = (const float*)d_in[2];
    const int*   in1 = (const int*)d_in[3];
    const int*   ou1 = (const int*)d_in[4];
    const int*   in2 = (const int*)d_in[5];
    const int*   ou2 = (const int*)d_in[6];
    float* out = (float*)d_out;

    const long NC = in_sizes[0];                 // N * 64
    const int  N  = (int)(NC / C_CH);
    const int  KK = in_sizes[1] / (C_CH * C_CH); // 27

    // workspace layout (256B aligned)
    size_t off = 0;
    auto take = [&](size_t bytes) { size_t o = off; off = (off + bytes + 255) & ~(size_t)255; return o; };
    char* ws = (char*)d_ws;
    bf16_t* xb    = (bf16_t*)(ws + take((size_t)NC * 2));     // x in bf16
    bf16_t* yb    = (bf16_t*)(ws + take((size_t)NC * 2));     // relu(norm(conv1)) in bf16
    float*  accb  = (float*)(ws + take((size_t)NC * 4));      // conv accumulator (reused)
    bf16_t* wf1   = (bf16_t*)(ws + take((size_t)KK * 4096 * 2));
    bf16_t* wf2   = (bf16_t*)(ws + take((size_t)KK * 4096 * 2));
    float*  stats = (float*)(ws + take(128 * 4));
    float*  sb    = (float*)(ws + take(128 * 4));

    const long n4 = NC / 4;
    const int  TPK = (N + 15) / 16;
    const int  ewBlocks = (int)((n4 + 255) / 256);
    const int  wprepN = KK * 4096;
    dim3 convGrid((TPK + 4 * TPW - 1) / (4 * TPW), KK);

    // --- prep: bf16 copies, weight fragments, zero accum + stats ---
    cvt_f32_bf16<<<ewBlocks, 256, 0, stream>>>(x, xb, n4);
    wprep_frag<<<(wprepN + 255) / 256, 256, 0, stream>>>(W1, wf1, wprepN);
    wprep_frag<<<(wprepN + 255) / 256, 256, 0, stream>>>(W2, wf2, wprepN);
    fill0_f4<<<2048, 256, 0, stream>>>(accb, n4);
    fill0_f4<<<1, 32, 0, stream>>>(stats, 32);

    // --- conv1 (dilation 1 map) ---
    sconv_wmma_bf16<<<convGrid, 128, 0, stream>>>(xb, wf1, in1, ou1, accb, N, TPK);

    // --- instance norm 1 + relu -> bf16 input for conv2 ---
    in_stats<<<1024, 256, 0, stream>>>(accb, stats, N);
    in_finalize<<<1, 64, 0, stream>>>(stats, sb, N);
    apply_norm_relu_bf16<<<ewBlocks, 256, 0, stream>>>(accb, sb, yb, n4);

    // --- conv2 (dilation 3 map) ---
    fill0_f4<<<2048, 256, 0, stream>>>(accb, n4);
    fill0_f4<<<1, 32, 0, stream>>>(stats, 32);
    sconv_wmma_bf16<<<convGrid, 128, 0, stream>>>(yb, wf2, in2, ou2, accb, N, TPK);

    // --- instance norm 2 + residual + relu -> fp32 output ---
    in_stats<<<1024, 256, 0, stream>>>(accb, stats, N);
    in_finalize<<<1, 64, 0, stream>>>(stats, sb, N);
    apply_norm_res_relu<<<ewBlocks, 256, 0, stream>>>(accb, sb, x, out, n4);
}
